// T5Attention_45346264711208
// MI455X (gfx1250) — compile-verified, matrix operand order checked by reading, not run on hardware
//
#include <hip/hip_runtime.h>
#include <hip/hip_bf16.h>

// ---------------- CDNA5 (gfx1250) T5 attention, bf16 WMMA + async-LDS ----------------
// Stages:
//   1) f32->bf16 convert of hidden_states; transpose+convert of wq/wk/wv/wo (scale folded into wq)
//   2) relative-position bias lookup table  bias[h][delta+2047]
//   3) QKV projections: 128x128 bf16 WMMA GEMM, async double-buffered LDS
//   4) flash attention (online softmax), WMMA for QK^T and PV
//   5) output projection GEMM -> f32 d_out

typedef __attribute__((ext_vector_type(16))) __bf16 v16bf;
typedef __attribute__((ext_vector_type(8)))  float  v8f;
typedef int v4i_ __attribute__((vector_size(16)));

#define NHEADS 16
#define DKV    64
#define HID    1024
#define SEQ    2048
#define BATCH  4

// ---- gfx1250 async global->LDS copy (ASYNCcnt path), guarded by __has_builtin ----
#if defined(__has_builtin)
#if __has_builtin(__builtin_amdgcn_global_load_async_to_lds_b128)
#define HAS_ASYNC_CP 1
#endif
#if __has_builtin(__builtin_amdgcn_s_wait_asynccnt)
#define HAS_WAIT_ASYNC 1
#endif
#endif

__device__ __forceinline__ void async_cp16(void* ldst, const void* gsrc) {
#ifdef HAS_ASYNC_CP
  __builtin_amdgcn_global_load_async_to_lds_b128(
      (__attribute__((address_space(1))) v4i_*)(void*)gsrc,
      (__attribute__((address_space(3))) v4i_*)ldst, 0, 0);
#else
  *(float4*)ldst = *(const float4*)gsrc;   // synchronous fallback
#endif
}

#ifdef HAS_ASYNC_CP
#ifdef HAS_WAIT_ASYNC
#define WAIT_ASYNC(n) __builtin_amdgcn_s_wait_asynccnt(n)
#else
#define WAIT_ASYNC(n) asm volatile("s_wait_asynccnt %0" :: "i"(n) : "memory")
#endif
#else
#define WAIT_ASYNC(n) ((void)0)
#endif

union FragBF {
  v16bf  v;
  float4 f[2];
  __bf16 e[16];
};

// ---------------------------------------------------------------- converts
struct __align__(8) BF4 { __bf16 x, y, z, w; };

__global__ __launch_bounds__(256) void k_f32_to_bf16(const float* __restrict__ in,
                                                     __bf16* __restrict__ out) {
  long i = (long)blockIdx.x * 256 + threadIdx.x;      // processes 4 elems
  float4 v = ((const float4*)in)[i];
  BF4 o; o.x = (__bf16)v.x; o.y = (__bf16)v.y; o.z = (__bf16)v.z; o.w = (__bf16)v.w;
  ((BF4*)out)[i] = o;
}

// [K=1024][N=1024] f32 -> [N][K] bf16 (optionally scaled)
__global__ __launch_bounds__(256) void k_transpose_bf16(const float* __restrict__ in,
                                                        __bf16* __restrict__ out,
                                                        float scale) {
  __shared__ float tile[32][33];
  int x  = blockIdx.x * 32 + threadIdx.x;   // source col (n)
  int y0 = blockIdx.y * 32;                 // source row (k)
  for (int i = threadIdx.y; i < 32; i += 8)
    tile[i][threadIdx.x] = in[(size_t)(y0 + i) * HID + x];
  __syncthreads();
  int nx  = blockIdx.y * 32 + threadIdx.x;  // dest col (k)
  int ny0 = blockIdx.x * 32;                // dest row (n)
  for (int i = threadIdx.y; i < 32; i += 8)
    out[(size_t)(ny0 + i) * HID + nx] = (__bf16)(tile[threadIdx.x][i] * scale);
}

// ---------------------------------------------------------------- T5 bias table
// bias_tab[h*4096 + (delta + 2047)] = rel_emb[bucket(delta)][h]
__global__ __launch_bounds__(256) void k_build_bias(const float* __restrict__ rel_emb,
                                                    float* __restrict__ bias_tab) {
  int idx = blockIdx.x * 256 + threadIdx.x;   // 16 * 4096
  int h = idx >> 12, d = idx & 4095;
  float val = 0.f;
  if (d < 4095) {
    int rp = d - 2047;                         // relative_position = k - q
    int bucket = (rp > 0) ? 16 : 0;
    int arp = rp < 0 ? -rp : rp;
    if (arp < 8) bucket += arp;
    else {
      float lf = 8.0f + __logf((float)arp * 0.125f) * (8.0f / __logf(16.0f));
      int bi = (int)lf; if (bi > 15) bi = 15;
      bucket += bi;
    }
    val = rel_emb[bucket * NHEADS + h];
  }
  bias_tab[idx] = val;
}

// ---------------------------------------------------------------- 128x128 WMMA GEMM
// C[M,N] = A[M,K] * Bt[N,K]^T ; bf16 inputs, f32 accumulate, bf16 or f32 output.
template <int OUT_F32>
__global__ __launch_bounds__(256) void k_gemm_bf16(const __bf16* __restrict__ A,
                                                   const __bf16* __restrict__ Bt,
                                                   void* __restrict__ C,
                                                   int M, int N, int K) {
  __shared__ __align__(16) __bf16 sA[2][128 * 32];
  __shared__ __align__(16) __bf16 sB[2][128 * 32];
  const int t = threadIdx.x, lane = t & 31, w = t >> 5;
  const int g = lane >> 4, lm = lane & 15;
  const int wm = w >> 1, wn = w & 1;                 // 4x2 wave grid -> 32x64 per wave
  const int mbase = blockIdx.x * 128, nbase = blockIdx.y * 128;

  v8f acc[2][4];
  for (int a = 0; a < 2; ++a)
    for (int b = 0; b < 4; ++b) acc[a][b] = (v8f){0,0,0,0,0,0,0,0};

  auto issue = [&](int kt, int buf) {
    for (int i = 0; i < 2; ++i) {
      int idx = t + i * 256;                   // 512 16B segments per tile
      int row = idx >> 2, seg = idx & 3;       // 32 halfs/row = 4 segs
      async_cp16(&sA[buf][row * 32 + seg * 8], A  + (size_t)(mbase + row) * K + kt + seg * 8);
      async_cp16(&sB[buf][row * 32 + seg * 8], Bt + (size_t)(nbase + row) * K + kt + seg * 8);
    }
  };

  issue(0, 0);
  const int KT = K >> 5;
  for (int it = 0; it < KT; ++it) {
    int buf = it & 1;
    if (it + 1 < KT) issue((it + 1) << 5, buf ^ 1);
    WAIT_ASYNC(4);                 // previous tile's copies done (4 in flight = next tile)
    __syncthreads();

    FragBF af[2], bfr[4];
    for (int mt = 0; mt < 2; ++mt) {
      const __bf16* p = &sA[buf][(wm * 32 + mt * 16 + lm) * 32];
      af[mt].f[0] = *(const float4*)(p + 8 * g);
      af[mt].f[1] = *(const float4*)(p + 16 + 8 * g);
    }
    for (int nt = 0; nt < 4; ++nt) {
      const __bf16* p = &sB[buf][(wn * 64 + nt * 16 + lm) * 32];
      bfr[nt].f[0] = *(const float4*)(p + 8 * g);
      bfr[nt].f[1] = *(const float4*)(p + 16 + 8 * g);
    }
    for (int mt = 0; mt < 2; ++mt)
      for (int nt = 0; nt < 4; ++nt)
        acc[mt][nt] = __builtin_amdgcn_wmma_f32_16x16x32_bf16(
            false, af[mt].v, false, bfr[nt].v, (short)0, acc[mt][nt], false, false);
    __syncthreads();
  }

  for (int mt = 0; mt < 2; ++mt)
    for (int nt = 0; nt < 4; ++nt)
      for (int j = 0; j < 8; ++j) {
        size_t row = mbase + wm * 32 + mt * 16 + j + 8 * g;   // C/D layout: M = j + 8g
        size_t col = nbase + wn * 64 + nt * 16 + lm;          // N = lane%16
        if (OUT_F32) ((float*)C)[row * (size_t)N + col] = acc[mt][nt][j];
        else         ((__bf16*)C)[row * (size_t)N + col] = (__bf16)acc[mt][nt][j];
      }
}

// ---------------------------------------------------------------- flash attention
// grid (S/64, heads, B); 128 threads = 4 waves; wave = 16 q rows, K-chunk = 32.
__global__ __launch_bounds__(128) void k_flash_attn(const __bf16* __restrict__ Q,
                                                    const __bf16* __restrict__ Kc,
                                                    const __bf16* __restrict__ Vc,
                                                    const float* __restrict__ bias_tab,
                                                    const unsigned char* __restrict__ mask,
                                                    __bf16* __restrict__ O) {
  __shared__ __align__(16) __bf16 sK[2][32 * 64];
  __shared__ __align__(16) __bf16 sV[2][32 * 64];
  __shared__ __align__(16) __bf16 sP[4][16 * 32];

  const int t = threadIdx.x, lane = t & 31, w = t >> 5;
  const int g = lane >> 4, lm = lane & 15;
  const int b = blockIdx.z, h = blockIdx.y, qt = blockIdx.x;
  const int q0 = qt * 64 + w * 16;
  const size_t tokbase = (size_t)b * SEQ;

  FragBF aq[2];     // Q A-frags, d = [0,32) and [32,64)  (scale already folded into wq)
  {
    const __bf16* qp = Q + (tokbase + q0 + lm) * HID + h * DKV;
    for (int dc = 0; dc < 2; ++dc) {
      aq[dc].f[0] = *(const float4*)(qp + dc * 32 + 8 * g);
      aq[dc].f[1] = *(const float4*)(qp + dc * 32 + 16 + 8 * g);
    }
  }

  float rmax[8], rsum[8];
  v8f accO[4];
  for (int j = 0; j < 8; ++j) { rmax[j] = -3.0e38f; rsum[j] = 0.f; }
  for (int d = 0; d < 4; ++d) accO[d] = (v8f){0,0,0,0,0,0,0,0};

  auto issue = [&](int it, int buf) {
    int k0 = it * 32;
    for (int i = 0; i < 2; ++i) {
      int idx = t + i * 128;                   // 256 segs each for K and V
      int row = idx >> 3, seg = idx & 7;       // 64 halfs/row = 8 segs
      size_t go = (tokbase + k0 + row) * HID + h * DKV + seg * 8;
      async_cp16(&sK[buf][row * 64 + seg * 8], Kc + go);
      async_cp16(&sV[buf][row * 64 + seg * 8], Vc + go);
    }
  };

  issue(0, 0);
  const int NIT = SEQ / 32;
  for (int it = 0; it < NIT; ++it) {
    int buf = it & 1;
    if (it + 1 < NIT) issue(it + 1, buf ^ 1);
    WAIT_ASYNC(4);
    __syncthreads();

    const int k0 = it * 32;
    float sc[2][8];
    for (int nc = 0; nc < 2; ++nc) {           // two 16-wide score tiles
      v8f s = (v8f){0,0,0,0,0,0,0,0};
      for (int dc = 0; dc < 2; ++dc) {
        FragBF bk;                             // B = K^T : lane n = k-token, runs over d
        const __bf16* p = &sK[buf][(nc * 16 + lm) * 64 + dc * 32];
        bk.f[0] = *(const float4*)(p + 8 * g);
        bk.f[1] = *(const float4*)(p + 16 + 8 * g);
        s = __builtin_amdgcn_wmma_f32_16x16x32_bf16(false, aq[dc].v, false, bk.v,
                                                    (short)0, s, false, false);
      }
      int kcol = k0 + nc * 16 + lm;
      float mb = mask[tokbase + kcol] ? 0.f : -1.0e30f;
      const float* bt = bias_tab + h * 4096 + kcol + 2047;
      for (int j = 0; j < 8; ++j) {
        int qrow = q0 + j + 8 * g;
        sc[nc][j] = s[j] + mb + bt[-qrow];     // bias[h][kcol - qrow + 2047]
      }
    }

    float pr[2][8];
    for (int j = 0; j < 8; ++j) {              // online softmax, row = q0 + j + 8g
      float cm = fmaxf(sc[0][j], sc[1][j]);
      for (int off = 1; off < 16; off <<= 1) cm = fmaxf(cm, __shfl_xor(cm, off, 32));
      float mnew  = fmaxf(rmax[j], cm);
      float scale = __expf(rmax[j] - mnew);
      float p0 = __expf(sc[0][j] - mnew);
      float p1 = __expf(sc[1][j] - mnew);
      pr[0][j] = p0; pr[1][j] = p1;
      float ps = p0 + p1;
      for (int off = 1; off < 16; off <<= 1) ps += __shfl_xor(ps, off, 32);
      rsum[j] = rsum[j] * scale + ps;
      rmax[j] = mnew;
      for (int d = 0; d < 4; ++d) accO[d][j] *= scale;
    }

    // transpose P (C/D layout -> A layout) through per-wave LDS scratch
    for (int nc = 0; nc < 2; ++nc)
      for (int j = 0; j < 8; ++j)
        sP[w][(j + 8 * g) * 32 + nc * 16 + lm] = (__bf16)pr[nc][j];
    __syncthreads();

    FragBF pf;
    pf.f[0] = *(const float4*)&sP[w][lm * 32 + 8 * g];
    pf.f[1] = *(const float4*)&sP[w][lm * 32 + 16 + 8 * g];

    for (int d = 0; d < 4; ++d) {              // O += P(16x32) * V(32x16 per d-block)
      FragBF vv;                               // B layout: lane n = d col, K pairs per elem
      for (int e = 0; e < 16; ++e) {
        int jj = e >> 1, hh = e & 1;
        int kp = (jj < 4) ? (2 * jj + hh + 8 * g) : (16 + 2 * (jj - 4) + hh + 8 * g);
        vv.e[e] = sV[buf][kp * 64 + d * 16 + lm];
      }
      accO[d] = __builtin_amdgcn_wmma_f32_16x16x32_bf16(false, pf.v, false, vv.v,
                                                        (short)0, accO[d], false, false);
    }
    __syncthreads();
  }

  for (int d = 0; d < 4; ++d)
    for (int j = 0; j < 8; ++j) {
      float val = accO[d][j] / rsum[j];
      size_t row = tokbase + q0 + j + 8 * g;
      O[row * HID + h * DKV + d * 16 + lm] = (__bf16)val;
    }
}

// ---------------------------------------------------------------- launch
extern "C" void kernel_launch(void* const* d_in, const int* in_sizes, int n_in,
                              void* d_out, int out_size, void* d_ws, size_t ws_size,
                              hipStream_t stream) {
  const float*         hs      = (const float*)d_in[0];
  const unsigned char* mask    = (const unsigned char*)d_in[1];
  const float*         wq      = (const float*)d_in[2];
  const float*         wk      = (const float*)d_in[3];
  const float*         wv      = (const float*)d_in[4];
  const float*         wo      = (const float*)d_in[5];
  const float*         rel_emb = (const float*)d_in[6];

  char* ws = (char*)d_ws;
  const size_t MB = 1u << 20;
  __bf16* hsb  = (__bf16*)(ws);             // 16 MB  [8192][1024]
  __bf16* wqt  = (__bf16*)(ws + 16 * MB);   //  2 MB  [N][K], scaled by 1/8
  __bf16* wkt  = (__bf16*)(ws + 18 * MB);
  __bf16* wvt  = (__bf16*)(ws + 20 * MB);
  __bf16* wot  = (__bf16*)(ws + 22 * MB);
  __bf16* qb   = (__bf16*)(ws + 24 * MB);   // 16 MB each
  __bf16* kb   = (__bf16*)(ws + 40 * MB);
  __bf16* vb   = (__bf16*)(ws + 56 * MB);
  __bf16* ab   = (__bf16*)(ws + 72 * MB);
  float*  btab = (float*)(ws + 88 * MB);    // 256 KB

  const int M = BATCH * SEQ;                // 8192

  // stage 1: converts
  k_f32_to_bf16<<<(M * HID) / (256 * 4), 256, 0, stream>>>(hs, hsb);
  dim3 tb(32, 8), tg(32, 32);
  k_transpose_bf16<<<tg, tb, 0, stream>>>(wq, wqt, 0.125f);   // fold 1/sqrt(64)
  k_transpose_bf16<<<tg, tb, 0, stream>>>(wk, wkt, 1.0f);
  k_transpose_bf16<<<tg, tb, 0, stream>>>(wv, wvt, 1.0f);
  k_transpose_bf16<<<tg, tb, 0, stream>>>(wo, wot, 1.0f);

  // stage 2: bias table
  k_build_bias<<<(NHEADS * 4096) / 256, 256, 0, stream>>>(rel_emb, btab);

  // stage 3: QKV projections
  dim3 gg(M / 128, HID / 128);
  k_gemm_bf16<0><<<gg, 256, 0, stream>>>(hsb, wqt, qb, M, HID, HID);
  k_gemm_bf16<0><<<gg, 256, 0, stream>>>(hsb, wkt, kb, M, HID, HID);
  k_gemm_bf16<0><<<gg, 256, 0, stream>>>(hsb, wvt, vb, M, HID, HID);

  // stage 4: flash attention
  k_flash_attn<<<dim3(SEQ / 64, NHEADS, BATCH), 128, 0, stream>>>(qb, kb, vb, btab, mask, ab);

  // stage 5: output projection -> f32
  k_gemm_bf16<1><<<gg, 256, 0, stream>>>(ab, wot, d_out, M, HID, HID);
}
